// ReproducingKernelHilbertSpaceDecoder_33328946217837
// MI455X (gfx1250) — matrix operand order repaired and checked by reading, not compiled, not run on hardware
//
#include <hip/hip_runtime.h>
#include <hip/hip_bf16.h>

// ---------------------------------------------------------------------------
// RKHS decoder, fused:  out = exp(-0.5*||z-a||^2 / (l^2+1e-7)) @ alpha
// Pre-pass: f32 -> bf16 conversion (+ alpha pre-transpose, row norms) into ws.
// Main: TDM (tensor_load_to_lds) double-buffered staging, BF16 WMMA compute.
// Output stores are non-temporal (TH=NT) so the 96MB write stream does not
// evict the L2-resident bf16 operand arrays that TDM re-reads every block.
// ---------------------------------------------------------------------------

typedef __attribute__((ext_vector_type(16))) __bf16 v16bf;
typedef __attribute__((ext_vector_type(8)))  float  v8f;
typedef __attribute__((ext_vector_type(4)))  float  v4f;   // ext-vector (builtin-compatible)
typedef __attribute__((ext_vector_type(4)))  unsigned int v4u;
typedef __attribute__((ext_vector_type(8)))  unsigned int v8u;

#define N_ROWS   32768
#define LATENT   256
#define NANCH    4096
#define FDIM     768
#define M_BLK    64
#define A_BLK    64
#define NCHUNK   (NANCH / A_BLK)        // 64

// workspace layout (bytes)
#define WS_ZB    0u                      // [32768][256] bf16 = 16 MB
#define WS_AB    16777216u               // [4096][256]  bf16 = 2 MB
#define WS_ALT   18874368u               // [64][768][64] bf16 = 6 MB (per-chunk K-major)
#define WS_ZSQ   25165824u               // [32768] f32
#define WS_ASQ   25296896u               // [4096]  f32

// LDS layout (bytes): z tile + double-buffered anchor/alpha tiles + S tile
#define ZT_OFF   0u                      // [64][256] bf16 = 32768
#define AT_OFF   32768u                  // 2 x [64][256] bf16 = 65536
#define ALT_OFF  98304u                  // 2 x [768][64] bf16 = 196608
#define SB_OFF   294912u                 // [64][64] bf16 = 8192
#define SMEM_BYTES 303104u
#define AT_BUF   32768u
#define ALT_BUF  98304u

union FragU { v16bf v; uint4 q[2]; };

// A fragment (16x32 bf16, row-major [m][k]):
// lanes 0-15: K=[0..7],[16..23]; lanes 16-31: K=[8..15],[24..31].
__device__ __forceinline__ v16bf load_fragA(const __bf16* lds, int row,
                                            int stride, int k0, int lane) {
  const int r = row + (lane & 15);
  const int h = (lane >> 4) << 3;
  const __bf16* p = lds + r * stride + k0 + h;
  FragU f;
  f.q[0] = *(const uint4*)(p);
  f.q[1] = *(const uint4*)(p + 16);
  return f.v;
}

// B fragment (32x16 bf16, stored K-contiguous per column [n][k]):
// lanes 0-15: K=[0..7],[16..23]; lanes 16-31: K=[8..15],[24..31].
__device__ __forceinline__ v16bf load_fragB(const __bf16* lds, int col,
                                            int stride, int k0, int lane) {
  const int n = col + (lane & 15);
  const int h = (lane >> 4) << 3;
  const __bf16* p = lds + n * stride + k0 + h;
  FragU f;
  f.q[0] = *(const uint4*)(p);
  f.q[1] = *(const uint4*)(p + 16);
  return f.v;
}

// 1D contiguous TDM transfer: global -> LDS, nbytes must be a multiple of 8.
// D# group0/group1 per CDNA5 ISA: count=1, type=2, data_size=3 (8B),
// 1-row tensor with tile_dim0 == tensor_dim0 == nbytes/8.
// 2-operand asm form => descriptor groups 2/3 disabled (<=2D tensor).
__device__ __forceinline__ void tdm_load_1d(void* lds_ptr, const void* gptr,
                                            unsigned nbytes) {
  const unsigned lds_off = (unsigned)(unsigned long long)lds_ptr;
  const unsigned long long ga = (unsigned long long)gptr;
  const unsigned n8 = nbytes >> 3;
  v4u g0;
  g0.x = 1u;                                   // count=1 (valid), user mode
  g0.y = lds_off;                              // lds_addr (bytes)
  g0.z = (unsigned)ga;                         // global_addr[31:0]
  g0.w = (unsigned)(ga >> 32) | (2u << 30);    // global_addr[56:32] | type=2
  v8u g1;
  g1[0] = 3u << 16;                            // wg_mask=0, data_size=3 (8B)
  g1[1] = (n8 & 0xFFFFu) << 16;                // tensor_dim0[15:0]
  g1[2] = (n8 >> 16) | (1u << 16);             // tensor_dim0[31:16] | tensor_dim1=1
  g1[3] = (n8 & 0xFFFFu) << 16;                // tile_dim0
  g1[4] = 1u;                                  // tile_dim1=1, tile_dim2=0
  g1[5] = n8;                                  // tensor_dim0_stride[31:0]
  g1[6] = 0u;
  g1[7] = 0u;
  asm volatile("tensor_load_to_lds %0, %1" :: "s"(g0), "s"(g1) : "memory");
}

// ---------------- pre-pass kernels ----------------------------------------

// f32 rows -> bf16 rows (+ per-row squared norm). 8 threads per 256-elem row.
// Source is read once: non-temporal loads keep it out of L2.
__global__ void prep_rows_kernel(const float* __restrict__ src,
                                 __bf16* __restrict__ dst,
                                 float* __restrict__ sq, int nrows) {
  const int g = blockIdx.x * blockDim.x + threadIdx.x;
  const int row = g >> 3;
  if (row >= nrows) return;
  const int seg = (g & 7) * 32;
  const float* s = src + (size_t)row * LATENT + seg;
  __bf16* d = dst + (size_t)row * LATENT + seg;
  float ss = 0.0f;
#pragma unroll
  for (int i = 0; i < 32; i += 4) {
    v4f v = __builtin_nontemporal_load((const v4f*)(s + i));
    ss += v.x * v.x + v.y * v.y + v.z * v.z + v.w * v.w;
    d[i + 0] = (__bf16)v.x; d[i + 1] = (__bf16)v.y;
    d[i + 2] = (__bf16)v.z; d[i + 3] = (__bf16)v.w;
  }
  ss += __shfl_xor(ss, 1);
  ss += __shfl_xor(ss, 2);
  ss += __shfl_xor(ss, 4);
  if ((threadIdx.x & 7) == 0) sq[row] = ss;
}

// alpha [4096][768] f32 -> per-chunk K-major bf16 tiles [ch][f][a_in_chunk].
__global__ void prep_alphaT_kernel(const float* __restrict__ alpha,
                                   __bf16* __restrict__ alT) {
  const int g = blockIdx.x * blockDim.x + threadIdx.x;   // [ch][f][a], a fastest
  const int a  = g & 63;
  const int f  = (g >> 6) % FDIM;
  const int ch = g / (64 * FDIM);
  const float v = __builtin_nontemporal_load(
      alpha + (size_t)(ch * A_BLK + a) * FDIM + f);
  alT[g] = (__bf16)v;
}

// ---------------- main kernel ---------------------------------------------

__global__ __launch_bounds__(512, 1)
void rkhs_decoder_kernel(const __bf16* __restrict__ zb,
                         const __bf16* __restrict__ ab,
                         const __bf16* __restrict__ alT_g,
                         const float* __restrict__ zsq_g,
                         const float* __restrict__ asq_g,
                         const float* __restrict__ log_ls,
                         float* __restrict__ out) {
  extern __shared__ char smem[];
  __bf16* zt = (__bf16*)(smem + ZT_OFF);
  __bf16* sb = (__bf16*)(smem + SB_OFF);

  const int tid     = threadIdx.x;
  const int lane    = tid & 31;
  const int wave    = tid >> 5;
  const int wave_m  = wave >> 2;            // stage 1: 4x4 grid over S 64x64
  const int wave_n  = wave & 3;
  const int wave_m2 = wave >> 3;            // stage 2: 2x8 grid (rows x feats)
  const int wave_n2 = wave & 7;
  const int h8      = (lane >> 4) << 3;
  const int nl      = lane & 15;
  const int blk     = blockIdx.x;

  const float l    = __expf(log_ls[0]);
  const float ninv = -0.5f / (l * l + 1e-7f);

  // Prologue: TDM-stage z block + chunk 0 anchor/alpha tiles into buffer 0.
  if (wave == 0) {
    tdm_load_1d(smem + ZT_OFF, zb + (size_t)blk * M_BLK * LATENT, M_BLK * LATENT * 2);
    tdm_load_1d(smem + AT_OFF, ab, A_BLK * LATENT * 2);
    tdm_load_1d(smem + ALT_OFF, alT_g, FDIM * A_BLK * 2);
    __builtin_amdgcn_s_wait_tensorcnt(0);
  }
  __syncthreads();

  // Per-block-constant row norms for the stage-1 epilogue (L2-resident).
  float zs[8];
#pragma unroll
  for (int r = 0; r < 8; ++r)
    zs[r] = zsq_g[blk * M_BLK + wave_m * 16 + h8 + r];

  v8f acc0[6], acc1[6];
#pragma unroll
  for (int t = 0; t < 6; ++t) { acc0[t] = (v8f)(0.0f); acc1[t] = (v8f)(0.0f); }

  for (int ch = 0; ch < NCHUNK; ++ch) {
    const int buf = ch & 1;
    __bf16* at  = (__bf16*)(smem + AT_OFF  + (unsigned)buf * AT_BUF);
    __bf16* alT = (__bf16*)(smem + ALT_OFF + (unsigned)buf * ALT_BUF);

    // Kick off next chunk's TDM transfers into the other buffer.
    if (wave == 0 && ch + 1 < NCHUNK) {
      tdm_load_1d(smem + AT_OFF + (unsigned)(buf ^ 1) * AT_BUF,
                  ab + (size_t)(ch + 1) * A_BLK * LATENT, A_BLK * LATENT * 2);
      tdm_load_1d(smem + ALT_OFF + (unsigned)(buf ^ 1) * ALT_BUF,
                  alT_g + (size_t)(ch + 1) * FDIM * A_BLK, FDIM * A_BLK * 2);
    }
    const float aq = asq_g[ch * A_BLK + wave_n * 16 + nl];

    // ---- stage 1: S = z_blk @ anchors_chunk^T, one 16x16 tile per wave -----
    {
      v8f s = (v8f)(0.0f);
#pragma unroll
      for (int kc = 0; kc < LATENT; kc += 32) {
        v16bf a = load_fragA(zt, wave_m * 16, LATENT, kc, lane);
        v16bf b = load_fragB(at, wave_n * 16, LATENT, kc, lane);
        s = __builtin_amdgcn_wmma_f32_16x16x32_bf16(
                false, a, false, b, (short)0, s, false, false);
      }
#pragma unroll
      for (int r = 0; r < 8; ++r) {
        const int ml  = wave_m * 16 + r + h8;
        const int al_ = wave_n * 16 + nl;
        float d = fmaxf(zs[r] + aq - 2.0f * s[r], 0.0f);
        sb[ml * A_BLK + al_] = (__bf16)__expf(ninv * d);
      }
    }
    __syncthreads();

    // ---- stage 2: acc += K_tile(bf16) @ alpha_chunk(bf16), 2x6 per wave ----
#pragma unroll
    for (int kc = 0; kc < A_BLK; kc += 32) {
      v16bf a0 = load_fragA(sb, wave_m2 * 32,      A_BLK, kc, lane);
      v16bf a1 = load_fragA(sb, wave_m2 * 32 + 16, A_BLK, kc, lane);
#pragma unroll
      for (int t = 0; t < 6; ++t) {
        v16bf b = load_fragB(alT, wave_n2 * 96 + t * 16, A_BLK, kc, lane);
        acc0[t] = __builtin_amdgcn_wmma_f32_16x16x32_bf16(
                      false, a0, false, b, (short)0, acc0[t], false, false);
        acc1[t] = __builtin_amdgcn_wmma_f32_16x16x32_bf16(
                      false, a1, false, b, (short)0, acc1[t], false, false);
      }
    }

    // Next chunk's tiles must be resident; issuing wave waits, then barrier
    // publishes the TDM-written LDS (and closes out sb/alT reads).
    if (wave == 0) __builtin_amdgcn_s_wait_tensorcnt(0);
    __syncthreads();
  }

  // ---- write out: f32 [32768][768], non-temporal (write-once stream) ------
#pragma unroll
  for (int t = 0; t < 6; ++t) {
    const int fcol = wave_n2 * 96 + t * 16 + nl;
#pragma unroll
    for (int r = 0; r < 8; ++r) {
      const int mrow = blk * M_BLK + wave_m2 * 32 + r + h8;
      __builtin_nontemporal_store(acc0[t][r], out + (size_t)mrow * FDIM + fcol);
      __builtin_nontemporal_store(acc1[t][r], out + (size_t)(mrow + 16) * FDIM + fcol);
    }
  }
}

extern "C" void kernel_launch(void* const* d_in, const int* in_sizes, int n_in,
                              void* d_out, int out_size, void* d_ws, size_t ws_size,
                              hipStream_t stream) {
  (void)in_sizes; (void)n_in; (void)ws_size; (void)out_size;
  const float* z       = (const float*)d_in[0];
  const float* anchors = (const float*)d_in[1];
  const float* alpha   = (const float*)d_in[2];
  const float* log_ls  = (const float*)d_in[3];
  float* out = (float*)d_out;
  char* ws = (char*)d_ws;

  __bf16* zb    = (__bf16*)(ws + WS_ZB);
  __bf16* ab    = (__bf16*)(ws + WS_AB);
  __bf16* alT_g = (__bf16*)(ws + WS_ALT);
  float*  zsq_g = (float*)(ws + WS_ZSQ);
  float*  asq_g = (float*)(ws + WS_ASQ);

  // pre-pass: bf16 conversion + row norms + alpha per-chunk transpose
  hipLaunchKernelGGL(prep_rows_kernel, dim3((N_ROWS * 8) / 512), dim3(512), 0,
                     stream, z, zb, zsq_g, N_ROWS);
  hipLaunchKernelGGL(prep_rows_kernel, dim3((NANCH * 8) / 512), dim3(512), 0,
                     stream, anchors, ab, asq_g, NANCH);
  hipLaunchKernelGGL(prep_alphaT_kernel, dim3((NANCH * FDIM) / 256), dim3(256),
                     0, stream, alpha, alT_g);

  // main fused kernel
  hipLaunchKernelGGL(rkhs_decoder_kernel, dim3(N_ROWS / M_BLK), dim3(512),
                     SMEM_BYTES, stream, zb, ab, alT_g, zsq_g, asq_g, log_ls, out);
}
